// RowColTransformer_51238959841540
// MI455X (gfx1250) — matrix operand order repaired and checked by reading, not compiled, not run on hardware
//
#include <hip/hip_runtime.h>

// ---------------------------------------------------------------------------
// Types for CDNA5 WMMA (wave32): V_WMMA_F32_16X16X32_BF16
// ---------------------------------------------------------------------------
typedef __bf16 bf16_t;
typedef bf16_t bf16x16 __attribute__((ext_vector_type(16)));
typedef float  f32x8   __attribute__((ext_vector_type(8)));

union BF16Frag { bf16x16 v; uint4 q[2]; };

// Model constants
#define NB      2048          // batch
#define NFEATS  32
#define DIMC    64
#define DROW    2048          // NFEATS*DIMC
#define INNER   512           // 8 heads * 64
#define TOK     65536L        // NB*NFEATS

// ---------------------------------------------------------------------------
// Fragment loaders (contiguous K order k = 16*hi + s, wave32)
// ---------------------------------------------------------------------------
// A-matrix 16x32 bf16: lane<16 holds row M=lane, K = {0..15};
//                      lane>=16 holds row M=lane-16, K = {16..31}
__device__ inline bf16x16 load_a_frag(const bf16_t* __restrict__ A, long lda,
                                      int m0, int k0, int lane) {
  int l = lane & 15, hi = lane >> 4;
  const bf16_t* p = A + (long)(m0 + l) * lda + k0 + hi * 16;
  BF16Frag f;
  f.q[0] = *(const uint4*)p;          // K offsets +0..7
  f.q[1] = *(const uint4*)(p + 8);    // K offsets +8..15
  return f.v;
}

// B-matrix 32x16 bf16 from PACKED weights: unit (kb, col) holds 32 contiguous
// K values for one output column. lane&15 = col, lane>=16 -> K+16.
__device__ inline bf16x16 load_b_packed(const bf16_t* __restrict__ Wp, int N,
                                        int k0, int n0, int lane) {
  int col = n0 + (lane & 15), hi = lane >> 4;
  const bf16_t* p = Wp + (((long)(k0 >> 5) * N + col) << 5) + (hi << 4);
  BF16Frag f;
  f.q[0] = *(const uint4*)p;
  f.q[1] = *(const uint4*)(p + 8);
  return f.v;
}

// B-matrix = K^T for attention scores: B[kk, j] = Kmat[j, kk] -> the 16 K
// values per lane are CONTIGUOUS in a row of Kmat.
__device__ inline bf16x16 load_bt_frag(const bf16_t* __restrict__ Kmat, long ld,
                                       int j0, int k0, int lane) {
  int j = j0 + (lane & 15), hi = lane >> 4;
  const bf16_t* p = Kmat + (long)j * ld + k0 + (hi << 4);
  BF16Frag f;
  f.q[0] = *(const uint4*)p;
  f.q[1] = *(const uint4*)(p + 8);
  return f.v;
}

// B-matrix from row-major V (strided): B[kk=j, col=d]
__device__ inline bf16x16 load_vb_frag(const bf16_t* __restrict__ V, long ld,
                                       int j0, int n0, int lane) {
  int col = n0 + (lane & 15), hi = lane >> 4;
  const bf16_t* p = V + (long)(j0 + hi * 16) * ld + col;
  bf16x16 f;
#pragma unroll
  for (int i = 0; i < 16; ++i) f[i] = p[(long)i * ld];
  return f;
}

__device__ inline f32x8 wmma_bf16(bf16x16 a, bf16x16 b, f32x8 c) {
  return __builtin_amdgcn_wmma_f32_16x16x32_bf16(false, a, false, b,
                                                 (short)0, c, false, false);
}

// ---------------------------------------------------------------------------
// Weight pack: f32 [nmat=4][K][N] -> bf16 packed B-fragment layout
//   dst[((kb*N + col)*32 + t)] = src[(kb*32 + t)*N + col]
// ---------------------------------------------------------------------------
__global__ void pack_w_kernel(const float* __restrict__ W, bf16_t* __restrict__ Wp,
                              int K, int N, long total) {
  long idx = (long)blockIdx.x * blockDim.x + threadIdx.x;
  if (idx >= total) return;
  long per = (long)K * N;
  long m = idx / per;
  long r = idx - m * per;
  long unit = r >> 5;
  int  t    = (int)(r & 31);
  long kb   = unit / N;
  int  col  = (int)(unit - kb * N);
  Wp[idx] = (bf16_t)W[m * per + (kb * 32 + t) * (long)N + col];
}

// ---------------------------------------------------------------------------
// Concat [B,16,64] + [B,16,64] -> X [B, 2048]
// ---------------------------------------------------------------------------
__global__ void concat_kernel(const float* __restrict__ x,
                              const float* __restrict__ xc,
                              float* __restrict__ X) {
  long idx = (long)blockIdx.x * blockDim.x + threadIdx.x; // over 2048*2048
  long row = idx >> 11;
  int  c   = (int)(idx & 2047);
  X[idx] = (c < 1024) ? x[row * 1024 + c] : xc[row * 1024 + (c - 1024)];
}

// ---------------------------------------------------------------------------
// LayerNorm: one block per row; writes f32 and bf16 copies
// ---------------------------------------------------------------------------
__global__ void layernorm_kernel(const float* __restrict__ x,
                                 const float* __restrict__ g,
                                 const float* __restrict__ beta,
                                 float* __restrict__ xn,
                                 bf16_t* __restrict__ xnh, int D) {
  long row = blockIdx.x;
  const float* xr = x + row * D;
  float s = 0.f, s2 = 0.f;
  for (int i = threadIdx.x; i < D; i += blockDim.x) {
    float v = xr[i]; s += v; s2 += v * v;
  }
#pragma unroll
  for (int d = 1; d < 32; d <<= 1) {
    s  += __shfl_xor(s, d, 32);
    s2 += __shfl_xor(s2, d, 32);
  }
  __shared__ float red[2][8];
  int wid = threadIdx.x >> 5;
  int nw  = blockDim.x >> 5;
  if ((threadIdx.x & 31) == 0) { red[0][wid] = s; red[1][wid] = s2; }
  __syncthreads();
  if (threadIdx.x == 0) {
    float a = 0.f, c = 0.f;
    for (int w = 0; w < nw; ++w) { a += red[0][w]; c += red[1][w]; }
    red[0][0] = a; red[1][0] = c;
  }
  __syncthreads();
  float mean = red[0][0] / D;
  float var  = red[1][0] / D - mean * mean;
  float inv  = rsqrtf(var + 1e-5f);
  for (int i = threadIdx.x; i < D; i += blockDim.x) {
    float v = (xr[i] - mean) * inv * g[i] + beta[i];
    xn[row * D + i]  = v;
    xnh[row * D + i] = (bf16_t)v;
  }
}

// ---------------------------------------------------------------------------
// GEMM: C[M,N] = A[M,K](bf16) @ Wpacked[K,N](bf16)  (+bias, +resid)
// block = 4 waves, each wave computes a 32x32 tile (2x2 WMMA C-tiles)
// ---------------------------------------------------------------------------
__global__ void __launch_bounds__(128)
gemm_bf16_kernel(const bf16_t* __restrict__ A, const bf16_t* __restrict__ Wp,
                 const float* __restrict__ bias, const float* __restrict__ resid,
                 float* __restrict__ outF, bf16_t* __restrict__ outH,
                 int M, int N, int K) {
  int lane = threadIdx.x & 31;
  int wave = threadIdx.x >> 5;
  int m0 = blockIdx.y * 64 + (wave >> 1) * 32;
  int n0 = blockIdx.x * 64 + (wave & 1) * 32;
  f32x8 acc[2][2] = {};
  for (int k0 = 0; k0 < K; k0 += 32) {
    bf16x16 a0 = load_a_frag(A, K, m0,      k0, lane);
    bf16x16 a1 = load_a_frag(A, K, m0 + 16, k0, lane);
    bf16x16 b0 = load_b_packed(Wp, N, k0, n0,      lane);
    bf16x16 b1 = load_b_packed(Wp, N, k0, n0 + 16, lane);
    acc[0][0] = wmma_bf16(a0, b0, acc[0][0]);
    acc[0][1] = wmma_bf16(a0, b1, acc[0][1]);
    acc[1][0] = wmma_bf16(a1, b0, acc[1][0]);
    acc[1][1] = wmma_bf16(a1, b1, acc[1][1]);
  }
  int l = lane & 15, hi = lane >> 4;
#pragma unroll
  for (int mi = 0; mi < 2; ++mi) {
#pragma unroll
    for (int ni = 0; ni < 2; ++ni) {
#pragma unroll
      for (int r = 0; r < 8; ++r) {
        int row = m0 + mi * 16 + hi * 8 + r;
        int col = n0 + ni * 16 + l;
        float v = acc[mi][ni][r];
        if (bias)  v += bias[col];
        if (resid) v += resid[(long)row * N + col];
        if (outF) outF[(long)row * N + col] = v;
        if (outH) outH[(long)row * N + col] = (bf16_t)v;
      }
    }
  }
}

// ---------------------------------------------------------------------------
// Fused flash attention, wave32, head_dim=64, 32-query block per wave.
// qkv: [batch*n, 1536] bf16 (Q|K|V, 8 heads x 64). out: [batch*n, 512] bf16.
// grid = (n/32, heads, batch), block = 32.
// ---------------------------------------------------------------------------
__global__ void __launch_bounds__(32)
flash_attn_kernel(const bf16_t* __restrict__ qkv, bf16_t* __restrict__ out, int n) {
  int lane = threadIdx.x;
  int l = lane & 15, hi = lane >> 4;
  int h = blockIdx.y, b = blockIdx.z;
  long base = (long)b * n * 1536;
  const bf16_t* Q  = qkv + base + h * 64;
  const bf16_t* Kp = qkv + base + 512 + h * 64;
  const bf16_t* Vp = qkv + base + 1024 + h * 64;
  int m0 = blockIdx.x * 32;

  bf16x16 qf[2][2];
#pragma unroll
  for (int mi = 0; mi < 2; ++mi)
#pragma unroll
    for (int ks = 0; ks < 2; ++ks)
      qf[mi][ks] = load_a_frag(Q, 1536, m0 + mi * 16, ks * 32, lane);

  f32x8 o[2][4] = {};
  float mrun[2][8], lrun[2][8];
#pragma unroll
  for (int mi = 0; mi < 2; ++mi)
#pragma unroll
    for (int r = 0; r < 8; ++r) { mrun[mi][r] = -1e30f; lrun[mi][r] = 0.f; }

  __shared__ float pS[32 * 32];
  const float scale = 0.125f; // 1/sqrt(64)

  for (int j0 = 0; j0 < n; j0 += 32) {
    // S = Q @ K^T  (32x32)
    f32x8 s[2][2] = {};
    bf16x16 kf[2][2];
#pragma unroll
    for (int nj = 0; nj < 2; ++nj)
#pragma unroll
      for (int ks = 0; ks < 2; ++ks)
        kf[nj][ks] = load_bt_frag(Kp, 1536, j0 + nj * 16, ks * 32, lane);
#pragma unroll
    for (int mi = 0; mi < 2; ++mi)
#pragma unroll
      for (int nj = 0; nj < 2; ++nj) {
        s[mi][nj] = wmma_bf16(qf[mi][0], kf[nj][0], s[mi][nj]);
        s[mi][nj] = wmma_bf16(qf[mi][1], kf[nj][1], s[mi][nj]);
      }

    // online softmax: rows live at (VGPR r, half 'hi'); j spans lanes&tiles
#pragma unroll
    for (int mi = 0; mi < 2; ++mi) {
#pragma unroll
      for (int r = 0; r < 8; ++r) {
        float s0 = s[mi][0][r] * scale;
        float s1 = s[mi][1][r] * scale;
        float mx = fmaxf(s0, s1);
#pragma unroll
        for (int d = 1; d < 16; d <<= 1) mx = fmaxf(mx, __shfl_xor(mx, d, 32));
        float mnew  = fmaxf(mrun[mi][r], mx);
        float alpha = __expf(mrun[mi][r] - mnew);
        float p0 = __expf(s0 - mnew);
        float p1 = __expf(s1 - mnew);
        s[mi][0][r] = p0;
        s[mi][1][r] = p1;
        float rs = p0 + p1;
#pragma unroll
        for (int d = 1; d < 16; d <<= 1) rs += __shfl_xor(rs, d, 32);
        lrun[mi][r] = lrun[mi][r] * alpha + rs;
        mrun[mi][r] = mnew;
#pragma unroll
        for (int nd = 0; nd < 4; ++nd) o[mi][nd][r] *= alpha;
      }
    }

    // P -> LDS (C layout) -> A-fragment layout (transpose across lanes)
#pragma unroll
    for (int mi = 0; mi < 2; ++mi)
#pragma unroll
      for (int nj = 0; nj < 2; ++nj)
#pragma unroll
        for (int r = 0; r < 8; ++r)
          pS[(mi * 16 + hi * 8 + r) * 32 + nj * 16 + l] = s[mi][nj][r];
    __syncthreads();
    bf16x16 pa[2];
#pragma unroll
    for (int mi = 0; mi < 2; ++mi) {
      bf16x16 f;
      int row = mi * 16 + l;
#pragma unroll
      for (int i = 0; i < 8; ++i) f[i]     = (bf16_t)pS[row * 32 + hi * 16 + i];
#pragma unroll
      for (int i = 0; i < 8; ++i) f[8 + i] = (bf16_t)pS[row * 32 + hi * 16 + 8 + i];
      pa[mi] = f;
    }

    // O += P @ V
#pragma unroll
    for (int nd = 0; nd < 4; ++nd) {
      bf16x16 vf = load_vb_frag(Vp, 1536, j0, nd * 16, lane);
      o[0][nd] = wmma_bf16(pa[0], vf, o[0][nd]);
      o[1][nd] = wmma_bf16(pa[1], vf, o[1][nd]);
    }
    __syncthreads();
  }

  // store O / l  -> out[token, h*64 + d]
#pragma unroll
  for (int mi = 0; mi < 2; ++mi)
#pragma unroll
    for (int nd = 0; nd < 4; ++nd)
#pragma unroll
      for (int r = 0; r < 8; ++r) {
        long row = (long)b * n + m0 + mi * 16 + hi * 8 + r;
        int  col = h * 64 + nd * 16 + l;
        out[row * 512 + col] = (bf16_t)(o[mi][nd][r] / lrun[mi][r]);
      }
}

// ---------------------------------------------------------------------------
// GEGLU: h [M, 2*DI] f32 -> out [M, DI] bf16 ; out = a * gelu_exact(g)
// ---------------------------------------------------------------------------
__global__ void geglu_kernel(const float* __restrict__ h, bf16_t* __restrict__ out,
                             int DI, long total) {
  long idx = (long)blockIdx.x * blockDim.x + threadIdx.x;
  if (idx >= total) return;
  long row = idx / DI;
  int  c   = (int)(idx - row * DI);
  float a  = h[row * 2L * DI + c];
  float gg = h[row * 2L * DI + DI + c];
  float ge = 0.5f * gg * (1.f + erff(gg * 0.70710678118f));
  out[idx] = (bf16_t)(a * ge);
}

// ---------------------------------------------------------------------------
// Host orchestration
// ---------------------------------------------------------------------------
extern "C" void kernel_launch(void* const* d_in, const int* in_sizes, int n_in,
                              void* d_out, int out_size, void* d_ws, size_t ws_size,
                              hipStream_t stream) {
  (void)in_sizes; (void)n_in; (void)ws_size;
  const float* in_x    = (const float*)d_in[0];
  const float* in_xc   = (const float*)d_in[1];
  const float* c_ln1_g = (const float*)d_in[2];
  const float* c_ln1_b = (const float*)d_in[3];
  const float* c_wqkv  = (const float*)d_in[4];
  const float* c_wo_w  = (const float*)d_in[5];
  const float* c_wo_b  = (const float*)d_in[6];
  const float* c_ln2_g = (const float*)d_in[7];
  const float* c_ln2_b = (const float*)d_in[8];
  const float* c_w1    = (const float*)d_in[9];
  const float* c_b1    = (const float*)d_in[10];
  const float* c_w2    = (const float*)d_in[11];
  const float* c_b2    = (const float*)d_in[12];
  const float* r_ln1_g = (const float*)d_in[13];
  const float* r_ln1_b = (const float*)d_in[14];
  const float* r_wqkv  = (const float*)d_in[15];
  const float* r_wo_w  = (const float*)d_in[16];
  const float* r_wo_b  = (const float*)d_in[17];
  const float* r_ln2_g = (const float*)d_in[18];
  const float* r_ln2_b = (const float*)d_in[19];
  const float* r_w1    = (const float*)d_in[20];
  const float* r_b1    = (const float*)d_in[21];
  const float* r_w2    = (const float*)d_in[22];
  const float* r_b2    = (const float*)d_in[23];

  char* ws = (char*)d_ws;
  size_t off = 0;
  auto take = [&](size_t bytes) -> char* {
    char* p = ws + off;
    off += (bytes + 255) & ~(size_t)255;
    return p;
  };

  float*  X    = (float*)take(TOK * 64 * 4);     // activations [2048,2048] f32
  float*  XN   = (float*)take(TOK * 64 * 4);     // LN output f32
  bf16_t* XNH  = (bf16_t*)take(TOK * 64 * 2);    // LN output bf16
  bf16_t* QKV  = (bf16_t*)take(TOK * 1536 * 2);  // qkv activations bf16
  bf16_t* ATT  = (bf16_t*)take(TOK * 512 * 2);   // attention output bf16
  float*  Hb   = (float*)take(TOK * 512 * 4);    // MLP hidden f32
  bf16_t* Gb   = (bf16_t*)take(TOK * 256 * 2);   // geglu output bf16
  bf16_t* Wc_qkv = (bf16_t*)take(4L * 64 * 1536 * 2);
  bf16_t* Wc_wo  = (bf16_t*)take(4L * 512 * 64 * 2);
  bf16_t* Wc_w1  = (bf16_t*)take(4L * 64 * 512 * 2);
  bf16_t* Wc_w2  = (bf16_t*)take(4L * 256 * 64 * 2);
  bf16_t* Wr_qkv = (bf16_t*)take(4L * 2048 * 1536 * 2);
  bf16_t* Wr_wo  = (bf16_t*)take(4L * 512 * 2048 * 2);
  bf16_t* Wr_w1  = (bf16_t*)take(4L * 2048 * 2048 * 2);
  bf16_t* Wr_w2  = (bf16_t*)take(4L * 1024 * 2048 * 2);

  auto pack = [&](const float* src, bf16_t* dst, int K, int N) {
    long total = 4L * K * N;
    pack_w_kernel<<<dim3((unsigned)((total + 255) / 256)), 256, 0, stream>>>(
        src, dst, K, N, total);
  };
  pack(c_wqkv, Wc_qkv, 64, 1536);
  pack(c_wo_w, Wc_wo, 512, 64);
  pack(c_w1,   Wc_w1, 64, 512);
  pack(c_w2,   Wc_w2, 256, 64);
  pack(r_wqkv, Wr_qkv, 2048, 1536);
  pack(r_wo_w, Wr_wo, 512, 2048);
  pack(r_w1,   Wr_w1, 2048, 2048);
  pack(r_w2,   Wr_w2, 1024, 2048);

  concat_kernel<<<dim3(16384), 256, 0, stream>>>(in_x, in_xc, X);

  auto gemm = [&](const bf16_t* A, const bf16_t* Wp, const float* bias,
                  const float* resid, float* oF, bf16_t* oH, int M, int N, int K) {
    gemm_bf16_kernel<<<dim3(N / 64, M / 64), 128, 0, stream>>>(
        A, Wp, bias, resid, oF, oH, M, N, K);
  };
  auto ln = [&](const float* xp, const float* g, const float* bb, int rows, int D) {
    layernorm_kernel<<<dim3((unsigned)rows), (D >= 256 ? 256 : D), 0, stream>>>(
        xp, g, bb, XN, XNH, D);
  };

  for (int d = 0; d < 4; ++d) {
    // ---------------- column block: tokens = 65536, width 64 ----------------
    ln(X, c_ln1_g + d * 64, c_ln1_b + d * 64, (int)TOK, 64);
    gemm(XNH, Wc_qkv + (long)d * 64 * 1536, nullptr, nullptr,
         nullptr, QKV, (int)TOK, 1536, 64);
    flash_attn_kernel<<<dim3(1, 8, NB), 32, 0, stream>>>(QKV, ATT, 32);
    gemm(ATT, Wc_wo + (long)d * 512 * 64, c_wo_b + d * 64, XN,
         X, nullptr, (int)TOK, 64, 512);
    ln(X, c_ln2_g + d * 64, c_ln2_b + d * 64, (int)TOK, 64);
    gemm(XNH, Wc_w1 + (long)d * 64 * 512, c_b1 + d * 512, nullptr,
         Hb, nullptr, (int)TOK, 512, 64);
    {
      long total = TOK * 256;
      geglu_kernel<<<dim3((unsigned)((total + 255) / 256)), 256, 0, stream>>>(
          Hb, Gb, 256, total);
    }
    gemm(Gb, Wc_w2 + (long)d * 256 * 64, c_b2 + d * 64, XN,
         X, nullptr, (int)TOK, 64, 256);

    // ---------------- row block: tokens = 2048, width 2048 ------------------
    ln(X, r_ln1_g + d * 2048, r_ln1_b + d * 2048, NB, 2048);
    gemm(XNH, Wr_qkv + (long)d * 2048 * 1536, nullptr, nullptr,
         nullptr, QKV, NB, 1536, 2048);
    flash_attn_kernel<<<dim3(64, 8, 1), 32, 0, stream>>>(QKV, ATT, 2048);
    gemm(ATT, Wr_wo + (long)d * 512 * 2048, r_wo_b + d * 2048, XN,
         X, nullptr, NB, 2048, 512);
    ln(X, r_ln2_g + d * 2048, r_ln2_b + d * 2048, NB, 2048);
    gemm(XNH, Wr_w1 + (long)d * 2048 * 2048, r_b1 + d * 2048, nullptr,
         Hb, nullptr, NB, 2048, 2048);
    {
      long total = (long)NB * 1024;
      geglu_kernel<<<dim3((unsigned)((total + 255) / 256)), 256, 0, stream>>>(
          Hb, Gb, 1024, total);
    }
    gemm(Gb, Wr_w2 + (long)d * 1024 * 2048, r_b2 + d * 2048, XN,
         X, nullptr, NB, 2048, 1024);
  }

  hipMemcpyAsync(d_out, X, (size_t)out_size * sizeof(float),
                 hipMemcpyDeviceToDevice, stream);
}
